// Attention_17179869184306
// MI455X (gfx1250) — matrix operand order repaired and checked by reading, not compiled, not run on hardware
//
#include <hip/hip_runtime.h>

// ---------------------------------------------------------------------------
// Types for CDNA5 WMMA (gfx1250, wave32)
// ---------------------------------------------------------------------------
typedef __attribute__((ext_vector_type(16))) __bf16        v16bf;
typedef __attribute__((ext_vector_type(8)))  float         v8f;
typedef __attribute__((ext_vector_type(4)))  unsigned int  u32x4;

union Frag {
  v16bf v;
  u32x4 q[2];
};

#define DEV static __device__ __forceinline__

DEV v8f wmma_bf16(Frag a, Frag b, v8f c) {
  // D = A(16x32 bf16) * B(32x16 bf16) + C(16x16 f32)
  return __builtin_amdgcn_wmma_f32_16x16x32_bf16(
      /*neg_a=*/false, a.v, /*neg_b=*/false, b.v,
      /*c_mod=*/(short)0, c, /*reuse_a=*/false, /*reuse_b=*/false);
}

constexpr int BB = 8;        // batch
constexpr int LL = 512;      // sequence length
constexpr int DD = 768;      // hidden
constexpr int HH = 12;       // heads
constexpr int HD = 64;       // head dim
constexpr float SCALE = 0.125f;   // 1/sqrt(64)
constexpr float NEGINF = -1e9f;

// ---------------------------------------------------------------------------
// Elementwise f32 -> bf16 conversion
// ---------------------------------------------------------------------------
__global__ void k_cvt_bf16(const float* __restrict__ src,
                           __bf16* __restrict__ dst, int n) {
  int i = blockIdx.x * blockDim.x + threadIdx.x;
  if (i < n) dst[i] = (__bf16)src[i];
}

// W[k][n] (f32, DxD) -> WT[n][k] (bf16, DxD)   (N-major "transposed-B" layout)
__global__ void k_transpose_cvt(const float* __restrict__ W,
                                __bf16* __restrict__ WT) {
  int n = blockIdx.x * 16 + threadIdx.x;
  int k = blockIdx.y * 16 + threadIdx.y;
  WT[(size_t)n * DD + k] = (__bf16)W[(size_t)k * DD + n];
}

// ---------------------------------------------------------------------------
// QKV projection.  grid = (4096/16, 12 heads, 3{q,k,v}), block = 32 (1 wave).
// Each wave: 16x64 output tile, 24 k-steps * 4 WMMAs = 96 v_wmma.
// Q,K written [B,H,L,HD] row-major; V written transposed [B,H,HD,L].
// ---------------------------------------------------------------------------
__global__ __launch_bounds__(32)
void k_qkv_gemm(const __bf16* __restrict__ Xb,
                const __bf16* __restrict__ WTbase,  // WqT|WkT|WvT contiguous
                const float* __restrict__ bq,
                const float* __restrict__ bk,
                const float* __restrict__ bv,
                __bf16* __restrict__ Qb,
                __bf16* __restrict__ Kb,
                __bf16* __restrict__ Vt) {
  const int lane = threadIdx.x;
  const int m  = lane & 15;     // row within tile (A) / column (B)
  const int hi = lane >> 4;     // lane-half
  const int m0 = blockIdx.x * 16;
  const int h  = blockIdx.y;
  const int which = blockIdx.z;            // 0=Q 1=K 2=V
  const int n0base = h * HD;

  const __bf16* wt = WTbase + (size_t)which * DD * DD;
  const float* bias = (which == 0) ? bq : (which == 1) ? bk : bv;

  v8f acc[4] = {};
  const __bf16* arow = Xb + (size_t)(m0 + m) * DD;

  for (int k0 = 0; k0 < DD; k0 += 32) {
    __builtin_prefetch(arow + k0 + 128, 0, 0);
    Frag a;
    a.q[0] = *(const u32x4*)(arow + k0 + 8 * hi);
    a.q[1] = *(const u32x4*)(arow + k0 + 16 + 8 * hi);
#pragma unroll
    for (int t = 0; t < 4; ++t) {
      const __bf16* bp = wt + (size_t)(n0base + t * 16 + m) * DD + k0 + 8 * hi;
      Frag bf;
      bf.q[0] = *(const u32x4*)(bp);
      bf.q[1] = *(const u32x4*)(bp + 16);
      acc[t] = wmma_bf16(a, bf, acc[t]);
    }
  }

#pragma unroll
  for (int t = 0; t < 4; ++t) {
    const int d = t * 16 + m;               // dim within head
    const float bv_ = bias[n0base + d];
#pragma unroll
    for (int r = 0; r < 8; ++r) {
      const int row = m0 + r + 8 * hi;      // global token index [0,4096)
      const int b = row >> 9;               // /512
      const int l = row & 511;
      const __bf16 val = (__bf16)(acc[t][r] + bv_);
      if (which == 2) {
        Vt[(((size_t)b * HH + h) * HD + d) * LL + l] = val;
      } else if (which == 0) {
        Qb[(((size_t)b * HH + h) * LL + l) * HD + d] = val;
      } else {
        Kb[(((size_t)b * HH + h) * LL + l) * HD + d] = val;
      }
    }
  }
}

// ---------------------------------------------------------------------------
// Fused attention: scores = QK^T/8 (+mask) -> softmax -> ctx = P@V.
// grid = (32 qtiles, 12 heads, 8 batch), block = 32 (1 wave).
// LDS: 16x512 f32 scores (32KB) + 16x512 bf16 probs (16KB) = 48KB.
// 64 + 64 = 128 v_wmma per wave.
// ---------------------------------------------------------------------------
__global__ __launch_bounds__(32)
void k_attn(const __bf16* __restrict__ Qb,
            const __bf16* __restrict__ Kb,
            const __bf16* __restrict__ Vt,
            const int* __restrict__ mask,
            __bf16* __restrict__ ctx) {
  __shared__ float s_s[16 * 512];
  __shared__ __align__(16) __bf16 s_p[16 * 512];

  const int lane = threadIdx.x;
  const int m  = lane & 15;
  const int hi = lane >> 4;
  const int q0 = blockIdx.x * 16;
  const int h  = blockIdx.y;
  const int b  = blockIdx.z;

  const size_t headQK = ((size_t)b * HH + h) * LL * HD;
  const size_t headV  = ((size_t)b * HH + h) * HD * LL;

  // Q A-fragments for k-steps 0 and 32 (HD = 64), held in registers all along.
  const __bf16* qrow = Qb + headQK + (size_t)(q0 + m) * HD;
  Frag aq0, aq1;
  aq0.q[0] = *(const u32x4*)(qrow + 8 * hi);
  aq0.q[1] = *(const u32x4*)(qrow + 16 + 8 * hi);
  aq1.q[0] = *(const u32x4*)(qrow + 32 + 8 * hi);
  aq1.q[1] = *(const u32x4*)(qrow + 48 + 8 * hi);

  // ---- Phase 1: scores row block (16 x 512) into LDS -----------------------
  for (int kt = 0; kt < 32; ++kt) {
    const int n0 = kt * 16;
    const __bf16* kp = Kb + headQK + (size_t)(n0 + m) * HD + 8 * hi;
    Frag b0, b1;
    b0.q[0] = *(const u32x4*)(kp);
    b0.q[1] = *(const u32x4*)(kp + 16);
    b1.q[0] = *(const u32x4*)(kp + 32);
    b1.q[1] = *(const u32x4*)(kp + 48);
    v8f acc = {};
    acc = wmma_bf16(aq0, b0, acc);
    acc = wmma_bf16(aq1, b1, acc);

    const int key = n0 + m;
    const bool live = mask[b * LL + key] > 0;
#pragma unroll
    for (int r = 0; r < 8; ++r) {
      float v = acc[r] * SCALE;
      if (!live) v = NEGINF;
      s_s[(r + 8 * hi) * 512 + key] = v;
    }
  }
  __syncthreads();

  // ---- Phase 2: softmax; two lanes cooperate per row -----------------------
  {
    const int row = lane >> 1;
    const int c0 = (lane & 1) * 256;
    float* sr = s_s + row * 512 + c0;
    float mx = -3.0e38f;
    for (int c = 0; c < 256; ++c) mx = fmaxf(mx, sr[c]);
    mx = fmaxf(mx, __shfl_xor(mx, 1));
    float sum = 0.f;
    for (int c = 0; c < 256; ++c) {
      float e = __expf(sr[c] - mx);
      sr[c] = e;
      sum += e;
    }
    sum += __shfl_xor(sum, 1);
    const float inv = 1.0f / sum;
    __bf16* pr = s_p + row * 512 + c0;
    for (int c = 0; c < 256; ++c) pr[c] = (__bf16)(sr[c] * inv);
  }
  __syncthreads();

  // ---- Phase 3: ctx(16x64) = P(16x512) @ V(512x64) -------------------------
  v8f acc[4] = {};
  for (int k0 = 0; k0 < 512; k0 += 32) {
    Frag a;
    const __bf16* pa = s_p + m * 512 + k0 + 8 * hi;
    a.q[0] = *(const u32x4*)(pa);
    a.q[1] = *(const u32x4*)(pa + 16);
#pragma unroll
    for (int t = 0; t < 4; ++t) {
      const __bf16* vp = Vt + headV + (size_t)(t * 16 + m) * LL + k0 + 8 * hi;
      Frag bf;
      bf.q[0] = *(const u32x4*)(vp);
      bf.q[1] = *(const u32x4*)(vp + 16);
      acc[t] = wmma_bf16(a, bf, acc[t]);
    }
  }
#pragma unroll
  for (int t = 0; t < 4; ++t) {
#pragma unroll
    for (int r = 0; r < 8; ++r) {
      ctx[((size_t)b * LL + q0 + r + 8 * hi) * DD + h * HD + t * 16 + m] =
          (__bf16)acc[t][r];
    }
  }
}

// ---------------------------------------------------------------------------
// Output projection + bias + residual.  grid = (256, 12), block = 32.
// ---------------------------------------------------------------------------
__global__ __launch_bounds__(32)
void k_oproj(const __bf16* __restrict__ ctxb,
             const __bf16* __restrict__ WoT,
             const float* __restrict__ bo,
             const float* __restrict__ x,
             float* __restrict__ res) {
  const int lane = threadIdx.x;
  const int m  = lane & 15;
  const int hi = lane >> 4;
  const int m0 = blockIdx.x * 16;
  const int n0base = blockIdx.y * 64;

  v8f acc[4] = {};
  const __bf16* arow = ctxb + (size_t)(m0 + m) * DD;
  for (int k0 = 0; k0 < DD; k0 += 32) {
    __builtin_prefetch(arow + k0 + 128, 0, 0);
    Frag a;
    a.q[0] = *(const u32x4*)(arow + k0 + 8 * hi);
    a.q[1] = *(const u32x4*)(arow + k0 + 16 + 8 * hi);
#pragma unroll
    for (int t = 0; t < 4; ++t) {
      const __bf16* bp = WoT + (size_t)(n0base + t * 16 + m) * DD + k0 + 8 * hi;
      Frag bf;
      bf.q[0] = *(const u32x4*)(bp);
      bf.q[1] = *(const u32x4*)(bp + 16);
      acc[t] = wmma_bf16(a, bf, acc[t]);
    }
  }
#pragma unroll
  for (int t = 0; t < 4; ++t) {
    const int n = n0base + t * 16 + m;
    const float bias = bo[n];
#pragma unroll
    for (int r = 0; r < 8; ++r) {
      const size_t row = m0 + r + 8 * hi;
      res[row * DD + n] = acc[t][r] + bias + x[row * DD + n];
    }
  }
}

// ---------------------------------------------------------------------------
// LayerNorm: one wave per row (4096 rows), 24 values/lane kept in registers.
// ---------------------------------------------------------------------------
__global__ __launch_bounds__(32)
void k_layernorm(const float* __restrict__ res,
                 const float* __restrict__ gamma,
                 const float* __restrict__ beta,
                 float* __restrict__ out) {
  const int row = blockIdx.x;
  const int lane = threadIdx.x;
  const float* r = res + (size_t)row * DD;
  float v[24];
#pragma unroll
  for (int i = 0; i < 24; ++i) v[i] = r[lane + i * 32];
  float s = 0.f;
#pragma unroll
  for (int i = 0; i < 24; ++i) s += v[i];
#pragma unroll
  for (int off = 16; off > 0; off >>= 1) s += __shfl_xor(s, off);
  const float mu = s * (1.0f / 768.0f);
  float s2 = 0.f;
#pragma unroll
  for (int i = 0; i < 24; ++i) { float d = v[i] - mu; s2 += d * d; }
#pragma unroll
  for (int off = 16; off > 0; off >>= 1) s2 += __shfl_xor(s2, off);
  const float inv = rsqrtf(s2 * (1.0f / 768.0f) + 1e-3f);
#pragma unroll
  for (int i = 0; i < 24; ++i) {
    const int c = lane + i * 32;
    out[(size_t)row * DD + c] = (v[i] - mu) * inv * gamma[c] + beta[c];
  }
}

// ---------------------------------------------------------------------------
// Host launcher
// ---------------------------------------------------------------------------
extern "C" void kernel_launch(void* const* d_in, const int* in_sizes, int n_in,
                              void* d_out, int out_size, void* d_ws, size_t ws_size,
                              hipStream_t stream) {
  (void)in_sizes; (void)n_in; (void)out_size; (void)ws_size;
  const float* x    = (const float*)d_in[0];
  const int*   mask = (const int*)d_in[1];
  const float* Wq   = (const float*)d_in[2];
  const float* bq   = (const float*)d_in[3];
  const float* Wk   = (const float*)d_in[4];
  const float* bk   = (const float*)d_in[5];
  const float* Wv   = (const float*)d_in[6];
  const float* bv   = (const float*)d_in[7];
  const float* Wo   = (const float*)d_in[8];
  const float* bo   = (const float*)d_in[9];
  const float* gamma = (const float*)d_in[10];
  const float* beta  = (const float*)d_in[11];
  float* out = (float*)d_out;

  const size_t NTOK = (size_t)BB * LL;         // 4096
  const size_t NX   = NTOK * DD;               // 3,145,728
  const size_t WSZ  = (size_t)DD * DD;         // 589,824

  char* ws = (char*)d_ws;
  size_t off = 0;
  __bf16* Xb  = (__bf16*)(ws + off); off += NX * 2;
  __bf16* WTb = (__bf16*)(ws + off); off += 4 * WSZ * 2;   // WqT|WkT|WvT|WoT
  __bf16* Qb  = (__bf16*)(ws + off); off += NX * 2;
  __bf16* Kb  = (__bf16*)(ws + off); off += NX * 2;
  __bf16* Vt  = (__bf16*)(ws + off); off += NX * 2;
  __bf16* Ctx = (__bf16*)(ws + off); off += NX * 2;
  float*  Res = (float*)(ws + off);  off += NX * 4;        // ~48.8 MB total

  // 1) x -> bf16
  k_cvt_bf16<<<dim3((unsigned)((NX + 255) / 256)), 256, 0, stream>>>(x, Xb, (int)NX);
  // 2) weight transposes (f32 -> bf16 N-major)
  dim3 tg(48, 48), tb(16, 16);
  k_transpose_cvt<<<tg, tb, 0, stream>>>(Wq, WTb + 0 * WSZ);
  k_transpose_cvt<<<tg, tb, 0, stream>>>(Wk, WTb + 1 * WSZ);
  k_transpose_cvt<<<tg, tb, 0, stream>>>(Wv, WTb + 2 * WSZ);
  k_transpose_cvt<<<tg, tb, 0, stream>>>(Wo, WTb + 3 * WSZ);
  // 3) QKV projection
  k_qkv_gemm<<<dim3(256, 12, 3), 32, 0, stream>>>(Xb, WTb, bq, bk, bv, Qb, Kb, Vt);
  // 4) fused attention
  k_attn<<<dim3(32, 12, 8), 32, 0, stream>>>(Qb, Kb, Vt, mask, Ctx);
  // 5) output projection + residual
  k_oproj<<<dim3(256, 12), 32, 0, stream>>>(Ctx, WTb + 3 * WSZ, bo, x, Res);
  // 6) layernorm
  k_layernorm<<<dim3(4096), 32, 0, stream>>>(Res, gamma, beta, out);
}
